// NEXT_Simulator_50921132262080
// MI455X (gfx1250) — compile-verified, matrix operand order checked by reading, not run on hardware
//
#include <hip/hip_runtime.h>
#include <hip/hip_bf16.h>

// ---------------- problem constants ----------------
#define NTICKS 550
#define NPMTS  12
#define SSI    2209          // 47*47
#define SPAD   2224          // padded to 139*16
#define NTILE4 139           // SPAD/16
#define BATCH  4
#define NELEC  8192
#define EPG    64            // electrons per workgroup
#define GAUSS_NORM 3.989422804f   // 1/(0.1*sqrt(2*pi))

typedef __attribute__((ext_vector_type(16))) _Float16 v16h;
typedef __attribute__((ext_vector_type(8)))  float    v8f;

// Fast sigmoid: v_exp_f32 + v_rcp_f32 (no IEEE div_scale/div_fixup chain).
__device__ __forceinline__ float sigmoidf(float x) {
    return __builtin_amdgcn_rcpf(1.0f + __expf(-x));
}

// A fragment: 16x32 f16 tile from LDS activations (row-major, stride in halves).
// lane%16 -> row, lane/16 selects which 16-wide K half.
__device__ __forceinline__ v16h load_a(const _Float16* lds, int stride, int mt, int k0, int lane) {
    int row = mt * 16 + (lane & 15);
    int kh  = k0 + (lane >> 4) * 16;
    return *(const v16h*)(lds + row * stride + kh);
}

// B fragment: pre-tiled global panel; one contiguous 32B load per lane.
__device__ __forceinline__ v16h load_b(const _Float16* panel, int nt, int kt, int KT, int lane) {
    size_t off = (((size_t)(nt * KT + kt) * 32) + lane) * 16;
    return *(const v16h*)(panel + off);
}

// One 16x16 output tile: A(LDS f16) x B(panel f16) over KT k-chunks, + bias, sigmoid, store f16 to LDS.
__device__ __forceinline__ void gemm_tile_sig(const _Float16* aLds, int aStride,
                                              const _Float16* bPanel, int KT,
                                              const float* bias,
                                              _Float16* outLds, int outStride,
                                              int mt, int nt, int lane) {
    v8f c = {0.f, 0.f, 0.f, 0.f, 0.f, 0.f, 0.f, 0.f};
    for (int kt = 0; kt < KT; ++kt) {
        v16h a = load_a(aLds, aStride, mt, kt * 32, lane);
        v16h b = load_b(bPanel, nt, kt, KT, lane);
        c = __builtin_amdgcn_wmma_f32_16x16x32_f16(false, a, false, b, (short)0, c, false, false);
    }
    int n = nt * 16 + (lane & 15);
    float bn = bias[n];
    int rbase = (lane >> 4) * 8;
#pragma unroll
    for (int r = 0; r < 8; ++r) {
        int m = mt * 16 + rbase + r;
        outLds[m * outStride + n] = (_Float16)sigmoidf(c[r] + bn);
    }
}

// ---------------- kernels ----------------

__global__ void zero_out(float* p, unsigned long long n) {
    unsigned long long i = (unsigned long long)blockIdx.x * blockDim.x + threadIdx.x;
    unsigned long long stride = (unsigned long long)gridDim.x * blockDim.x;
    for (; i < n; i += stride) p[i] = 0.0f;
}

// Convert Ws2/Ws3/Ws4 (f32 row-major KxN) into f16 WMMA B-panels in workspace.
// panel layout: idx = ((nt*KT + kt)*32 + lane)*16 + h ; lane%16 = col, lane/16 = K-half, h = k offset.
__global__ void prep_panels(const float* Ws2, const float* Ws3, const float* Ws4,
                            _Float16* p2, _Float16* p3, _Float16* p4) {
    int gid = blockIdx.x * blockDim.x + threadIdx.x;
    int stride = gridDim.x * blockDim.x;
    // layer2: K=64 (KT=2), N=128 (NT=8)
    for (int i = gid; i < 2 * 8 * 512; i += stride) {
        int h = i & 15, lane = (i >> 4) & 31, tile = i >> 9;
        int kt = tile % 2, nt = tile / 2;
        int k = kt * 32 + (lane >> 4) * 16 + h;
        int n = nt * 16 + (lane & 15);
        p2[i] = (_Float16)Ws2[k * 128 + n];
    }
    // layer3: K=128 (KT=4), N=256 (NT=16)
    for (int i = gid; i < 4 * 16 * 512; i += stride) {
        int h = i & 15, lane = (i >> 4) & 31, tile = i >> 9;
        int kt = tile % 4, nt = tile / 4;
        int k = kt * 32 + (lane >> 4) * 16 + h;
        int n = nt * 16 + (lane & 15);
        p3[i] = (_Float16)Ws3[k * 256 + n];
    }
    // layer4: K=256 (KT=8), N=2209 padded to 2224 (NT=139)
    for (int i = gid; i < 8 * NTILE4 * 512; i += stride) {
        int h = i & 15, lane = (i >> 4) & 31, tile = i >> 9;
        int kt = tile % 8, nt = tile / 8;
        int k = kt * 32 + (lane >> 4) * 16 + h;
        int n = nt * 16 + (lane & 15);
        p4[i] = (n < SSI) ? (_Float16)Ws4[k * SSI + n] : (_Float16)0.0f;
    }
}

__global__ void __launch_bounds__(256)
fused_kernel(const float* __restrict__ electrons, const float* __restrict__ weight,
             const float* __restrict__ Wp1, const float* __restrict__ bp1,
             const float* __restrict__ Wp2, const float* __restrict__ bp2,
             const float* __restrict__ pmt_scale,
             const float* __restrict__ Ws1, const float* __restrict__ bs1,
             const float* __restrict__ bs2, const float* __restrict__ bs3,
             const float* __restrict__ bs4, const float* __restrict__ si_scale,
             const _Float16* __restrict__ p2, const _Float16* __restrict__ p3,
             const _Float16* __restrict__ p4,
             float* __restrict__ out_pmt, float* __restrict__ out_sipm) {
    const int tid  = threadIdx.x;
    const int lane = tid & 31;
    const int wave = tid >> 5;
    const int b    = blockIdx.x / (NELEC / EPG);
    const int eg0  = (blockIdx.x % (NELEC / EPG)) * EPG;

    extern __shared__ char smem_raw[];
    _Float16* h1 = (_Float16*)smem_raw;          // 64 x 64   ( 8 KB)
    _Float16* h2 = h1 + 64 * 64;                 // 64 x 128  (16 KB)
    _Float16* h3 = h2 + 64 * 128;                // 64 x 256  (32 KB)
    float*   exw = (float*)(h3 + 64 * 256);      // 64 x 8    ( 2 KB)
    int*     t0s = (int*)(exw + 64 * 8);         // 64
    float* pmt_lds = (float*)(t0s + 64);         // 12 x 550  (26.4 KB)

    for (int i = tid; i < NPMTS * NTICKS; i += 256) pmt_lds[i] = 0.0f;

    // ---- stage 1: gaussian tick windows + layer 1 (K=2, scalar) ----
    {
        int e = tid >> 2;            // 0..63
        int part = tid & 3;
        size_t eg = (size_t)b * NELEC + (eg0 + e);
        float x = electrons[eg * 3 + 0];
        float y = electrons[eg * 3 + 1];
        float z = electrons[eg * 3 + 2];
        float w = weight[eg];
        if (part == 0) {
            int t0 = (int)floorf(z - 0.5f) - 3;
            t0s[e] = t0;
#pragma unroll
            for (int j = 0; j < 8; ++j) {
                float dt = (float)(t0 + j) + 0.5f - z;
                exw[e * 8 + j] = GAUSS_NORM * __expf(-10.0f * dt * dt) * w;
            }
        }
#pragma unroll 4
        for (int i = 0; i < 16; ++i) {
            int k = part * 16 + i;
            h1[e * 64 + k] = (_Float16)sigmoidf(x * Ws1[k] + y * Ws1[64 + k] + bs1[k]);
        }
    }
    __syncthreads();

    // ---- PMT path (tiny MLP + LDS scatter), threads 0..63 ----
    if (tid < EPG) {
        int e = tid;
        size_t eg = (size_t)b * NELEC + (eg0 + e);
        float x = electrons[eg * 3 + 0];
        float y = electrons[eg * 3 + 1];
        float hh[28];
#pragma unroll
        for (int i = 0; i < 28; ++i) hh[i] = sigmoidf(x * Wp1[i] + y * Wp1[28 + i] + bp1[i]);
        int t0 = t0s[e];
        const float* ex = exw + e * 8;
        for (int p = 0; p < NPMTS; ++p) {
            float acc = bp2[p];
#pragma unroll
            for (int i = 0; i < 28; ++i) acc += hh[i] * Wp2[i * NPMTS + p];
            float sc = pmt_scale[p];
            float resp = sigmoidf(acc) * sc * sc;
#pragma unroll
            for (int j = 0; j < 8; ++j) {
                int t = t0 + j;
                if ((unsigned)t < NTICKS) atomicAdd(&pmt_lds[p * NTICKS + t], resp * ex[j]);
            }
        }
    }

    // ---- layer 2: 64x64 @ 64x128 (WMMA), 32 tiles over 8 waves ----
    for (int idx = wave; idx < 32; idx += 8)
        gemm_tile_sig(h1, 64, p2, 2, bs2, h2, 128, idx >> 3, idx & 7, lane);
    __syncthreads();

    // flush PMT plane to global
    for (int i = tid; i < NPMTS * NTICKS; i += 256)
        atomicAdd(&out_pmt[(size_t)b * NPMTS * NTICKS + i], pmt_lds[i]);

    // ---- layer 3: 64x128 @ 128x256 (WMMA), 64 tiles ----
    for (int idx = wave; idx < 64; idx += 8)
        gemm_tile_sig(h2, 128, p3, 4, bs3, h3, 256, idx >> 4, idx & 15, lane);
    __syncthreads();

    // ---- layer 4: 64x256 @ 256x2224 (WMMA) fused with sparse tick scatter ----
    for (int idx = wave; idx < NTILE4 * 4; idx += 8) {
        int nt = idx >> 2, mt = idx & 3;

        // prefetch next tile's B panel (global_prefetch_b8; L2-resident stream)
        int nidx = idx + 8;
        if (nidx < NTILE4 * 4) {
            const _Float16* nb = p4 + (((size_t)((nidx >> 2) * 8) * 32) + lane) * 16;
            __builtin_prefetch(nb, 0, 3);
        }

        v8f c = {0.f, 0.f, 0.f, 0.f, 0.f, 0.f, 0.f, 0.f};
        for (int kt = 0; kt < 8; ++kt) {
            v16h a = load_a(h3, 256, mt, kt * 32, lane);
            v16h bf = load_b(p4, nt, kt, 8, lane);
            c = __builtin_amdgcn_wmma_f32_16x16x32_f16(false, a, false, bf, (short)0, c, false, false);
        }
        int s = nt * 16 + (lane & 15);
        if (s < SSI) {
            float bn  = bs4[s];
            float sc  = si_scale[s];
            float sc2 = sc * sc;
            float* outb = out_sipm + ((size_t)b * SSI + s) * NTICKS;
            int rbase = (lane >> 4) * 8;
#pragma unroll
            for (int r = 0; r < 8; ++r) {
                int e = mt * 16 + rbase + r;
                float resp = sigmoidf(c[r] + bn) * sc2;
                int t0 = t0s[e];
                const float* ex = exw + e * 8;
#pragma unroll
                for (int j = 0; j < 8; ++j) {
                    int t = t0 + j;
                    if ((unsigned)t < NTICKS) atomicAdd(outb + t, resp * ex[j]);
                }
            }
        }
    }
}

// ---------------- host launcher ----------------
extern "C" void kernel_launch(void* const* d_in, const int* in_sizes, int n_in,
                              void* d_out, int out_size, void* d_ws, size_t ws_size,
                              hipStream_t stream) {
    const float* electrons = (const float*)d_in[0];
    const float* weight    = (const float*)d_in[1];
    const float* Wp1       = (const float*)d_in[2];
    const float* bp1       = (const float*)d_in[3];
    const float* Wp2       = (const float*)d_in[4];
    const float* bp2       = (const float*)d_in[5];
    const float* pmt_scale = (const float*)d_in[6];
    const float* Ws1       = (const float*)d_in[7];
    const float* bs1       = (const float*)d_in[8];
    const float* Ws2       = (const float*)d_in[9];
    const float* bs2       = (const float*)d_in[10];
    const float* Ws3       = (const float*)d_in[11];
    const float* bs3       = (const float*)d_in[12];
    const float* Ws4       = (const float*)d_in[13];
    const float* bs4       = (const float*)d_in[14];
    const float* si_scale  = (const float*)d_in[15];

    _Float16* p2 = (_Float16*)d_ws;            //  8192 halves
    _Float16* p3 = p2 + 2 * 8 * 512;           // 32768 halves
    _Float16* p4 = p3 + 4 * 16 * 512;          // 569344 halves (~1.2 MB total)

    float* out_pmt  = (float*)d_out;
    float* out_sipm = out_pmt + (size_t)BATCH * NPMTS * NTICKS;

    zero_out<<<2048, 256, 0, stream>>>((float*)d_out, (unsigned long long)out_size);
    prep_panels<<<512, 256, 0, stream>>>(Ws2, Ws3, Ws4, p2, p3, p4);

    size_t lds_bytes = (64 * 64 + 64 * 128 + 64 * 256) * sizeof(_Float16)
                     + (64 * 8) * sizeof(float) + 64 * sizeof(int)
                     + NPMTS * NTICKS * sizeof(float);   // ~84 KB, fits CDNA5 320 KB LDS

    fused_kernel<<<dim3(BATCH * (NELEC / EPG)), 256, lds_bytes, stream>>>(
        electrons, weight, Wp1, bp1, Wp2, bp2, pmt_scale,
        Ws1, bs1, bs2, bs3, bs4, si_scale,
        p2, p3, p4, out_pmt, out_sipm);
}
